// SAMCMultiHeadAttention_40973988004727
// MI455X (gfx1250) — matrix operand order repaired and checked by reading, not compiled, not run on hardware
//
#include <hip/hip_runtime.h>
#include <hip/hip_bf16.h>
#include <stdint.h>

// Problem constants (match reference)
#define BB 2
#define SS 4096
#define DD 512
#define HH 8
#define DHD 64

typedef __attribute__((ext_vector_type(16))) __bf16 v16bf;
typedef __attribute__((ext_vector_type(8)))  float  v8f;

union Frag {
  v16bf v;
  uint4 q[2];
  unsigned short u[16];
};

// Hardware convert: let the backend pick v_cvt_*bf16_f32 (packed where possible)
__device__ __forceinline__ unsigned short f2bf(float f) {
  return __builtin_bit_cast(unsigned short, (__bf16)f);
}

__device__ __forceinline__ v8f wmma_bf16(const Frag& a, const Frag& b, v8f c) {
  return __builtin_amdgcn_wmma_f32_16x16x32_bf16(
      /*neg_a=*/false, a.v, /*neg_b=*/false, b.v,
      /*c_mod=*/(short)0, c, /*reuse_a=*/false, /*reuse_b=*/false);
}

// ---------------------------------------------------------------------------
// Kernel 0: bulk f32 -> bf16 conversion (8 elements / thread, vectorized)
// ---------------------------------------------------------------------------
__global__ __launch_bounds__(256)
void cvt_bf16_kernel(const float* __restrict__ src,
                     unsigned short* __restrict__ dst) {
  size_t i = ((size_t)blockIdx.x * blockDim.x + threadIdx.x) << 3;
  float4 f0 = *(const float4*)(src + i);
  float4 f1 = *(const float4*)(src + i + 4);
  union { unsigned short u[8]; uint4 v4; } o;
  o.u[0] = f2bf(f0.x); o.u[1] = f2bf(f0.y);
  o.u[2] = f2bf(f0.z); o.u[3] = f2bf(f0.w);
  o.u[4] = f2bf(f1.x); o.u[5] = f2bf(f1.y);
  o.u[6] = f2bf(f1.z); o.u[7] = f2bf(f1.w);
  *(uint4*)(dst + i) = o.v4;
}

// ---------------------------------------------------------------------------
// Kernel 1: QKV projection GEMM, all-bf16 operands.
// Each wave: 16 rows x 64 cols (= one full head) -> 4 accumulators,
// one A-frag load feeds 4 WMMAs per k-step.
// vt==0: store bf16 head-major [B,H,S,64]; vt==1: store transposed [B,H,64,S]
// ---------------------------------------------------------------------------
__global__ __launch_bounds__(256)
void proj_gemm_kernel(const unsigned short* __restrict__ A,   // [8192,512] bf16
                      const unsigned short* __restrict__ Wb,  // [512,512] bf16 (row = out col)
                      const float* __restrict__ bias,
                      unsigned short* __restrict__ dst, int vt) {
  int wid  = (blockIdx.x << 3) + (threadIdx.x >> 5);   // 0..4095
  int lane = threadIdx.x & 31;
  int m    = lane & 15;
  int half = lane >> 4;
  int rowt = wid >> 3;                                  // 0..511
  int cg   = wid & 7;                                   // head / 64-col group
  int r0   = rowt << 4;

  const unsigned short* ap = A + ((size_t)(r0 + m) << 9);

  v8f acc[4] = {{}, {}, {}, {}};
  for (int kb = 0; kb < DD; kb += 32) {
    Frag a;
    a.q[0] = *(const uint4*)(ap + kb + half * 16);
    a.q[1] = *(const uint4*)(ap + kb + half * 16 + 8);
#pragma unroll
    for (int t = 0; t < 4; ++t) {
      int col = (cg << 6) + (t << 4) + m;
      const unsigned short* wp = Wb + ((size_t)col << 9) + kb + half * 16;
      Frag b;
      b.q[0] = *(const uint4*)wp;
      b.q[1] = *(const uint4*)(wp + 8);
      acc[t] = wmma_bf16(a, b, acc[t]);
    }
  }

  int bidx = r0 >> 12;                                  // batch
  int s0   = r0 & (SS - 1);                             // seq base
#pragma unroll
  for (int t = 0; t < 4; ++t) {
    int col = (cg << 6) + (t << 4) + m;
    float bv = bias[col];
    int dh = (t << 4) + m;                              // head col (head == cg)
    if (!vt) {
#pragma unroll
      for (int r = 0; r < 8; ++r) {
        int s = s0 + r + (half << 3);
        dst[((size_t)((bidx * HH + cg) * SS + s) << 6) + dh] = f2bf(acc[t][r] + bv);
      }
    } else {
      union { unsigned short pk[8]; uint4 v4; } pu;
#pragma unroll
      for (int r = 0; r < 8; ++r) pu.pk[r] = f2bf(acc[t][r] + bv);
      size_t base = ((size_t)(bidx * HH + cg) * DHD + dh) * SS + s0 + (half << 3);
      *(uint4*)(dst + base) = pu.v4;
    }
  }
}

// ---------------------------------------------------------------------------
// Kernel 2: online-softmax stats.  Per wave: one 16-query tile of one (b,h).
// Computes row max m and inv_den = 1/(sum(exp(s-m)) + exp(const-m)*S).
// ---------------------------------------------------------------------------
__global__ __launch_bounds__(256)
void attn_stats_kernel(const unsigned short* __restrict__ Qh,
                       const unsigned short* __restrict__ Kh,
                       const float* __restrict__ cst,
                       float* __restrict__ Mrow, float* __restrict__ Drow) {
  int wid  = (blockIdx.x << 3) + (threadIdx.x >> 5);    // 0..4095
  int lane = threadIdx.x & 31;
  int m    = lane & 15;
  int half = lane >> 4;
  int bh   = wid >> 8;                                  // 0..15
  int q0   = (wid & 255) << 4;
  int h    = bh & 7;

  const unsigned short* qp    = Qh + (((size_t)bh * SS + q0 + m) << 6);
  const unsigned short* kbase = Kh + ((size_t)bh * SS << 6);

  Frag qa0, qa1;                                        // Q tile, dh 0..31 / 32..63
  qa0.q[0] = *(const uint4*)(qp + half * 16);
  qa0.q[1] = *(const uint4*)(qp + half * 16 + 8);
  qa1.q[0] = *(const uint4*)(qp + 32 + half * 16);
  qa1.q[1] = *(const uint4*)(qp + 32 + half * 16 + 8);

  float pm[8], ps[8];
#pragma unroll
  for (int r = 0; r < 8; ++r) { pm[r] = -3.0e38f; ps[r] = 0.0f; }

  for (int kb = 0; kb < SS; kb += 32) {
    const unsigned short* k0 = kbase + ((size_t)(kb + m) << 6);
    const unsigned short* k1 = kbase + ((size_t)(kb + 16 + m) << 6);
    __builtin_prefetch(kbase + ((size_t)((kb + 32) & (SS - 1)) << 6), 0, 3);
    Frag b00, b01, b10, b11;
    b00.q[0] = *(const uint4*)(k0 + half * 16);
    b00.q[1] = *(const uint4*)(k0 + half * 16 + 8);
    b01.q[0] = *(const uint4*)(k0 + 32 + half * 16);
    b01.q[1] = *(const uint4*)(k0 + 32 + half * 16 + 8);
    b10.q[0] = *(const uint4*)(k1 + half * 16);
    b10.q[1] = *(const uint4*)(k1 + half * 16 + 8);
    b11.q[0] = *(const uint4*)(k1 + 32 + half * 16);
    b11.q[1] = *(const uint4*)(k1 + 32 + half * 16 + 8);

    v8f c0 = {}; c0 = wmma_bf16(qa0, b00, c0); c0 = wmma_bf16(qa1, b01, c0);
    v8f c1 = {}; c1 = wmma_bf16(qa0, b10, c1); c1 = wmma_bf16(qa1, b11, c1);

#pragma unroll
    for (int r = 0; r < 8; ++r) {
      float s0v = c0[r] * 0.125f;                       // 1/sqrt(64)
      float s1v = c1[r] * 0.125f;
      float nm  = fmaxf(pm[r], fmaxf(s0v, s1v));
      ps[r] = ps[r] * __expf(pm[r] - nm) + __expf(s0v - nm) + __expf(s1v - nm);
      pm[r] = nm;
    }
  }

  // combine partials across the 16 lanes holding the same rows
#pragma unroll
  for (int mask = 1; mask < 16; mask <<= 1) {
#pragma unroll
    for (int r = 0; r < 8; ++r) {
      float om = __shfl_xor(pm[r], mask, 32);
      float os = __shfl_xor(ps[r], mask, 32);
      float nm = fmaxf(pm[r], om);
      ps[r] = ps[r] * __expf(pm[r] - nm) + os * __expf(om - nm);
      pm[r] = nm;
    }
  }

  if (m == 0) {
    float cc = cst[h];
#pragma unroll
    for (int r = 0; r < 8; ++r) {
      float leak = __expf(cc - pm[r]) * (float)SS;
      int row = q0 + r + (half << 3);
      Mrow[(size_t)bh * SS + row] = pm[r];
      Drow[(size_t)bh * SS + row] = 1.0f / (ps[r] + leak);
    }
  }
}

// ---------------------------------------------------------------------------
// Kernel 3: recompute scores, write normalized attn (f32), fuse attn @ V.
// attn tile transposed D->A layout through a per-wave LDS tile.
// ---------------------------------------------------------------------------
__global__ __launch_bounds__(256)
void attn_av_kernel(const unsigned short* __restrict__ Qh,
                    const unsigned short* __restrict__ Kh,
                    const unsigned short* __restrict__ Vt,
                    const float* __restrict__ Mrow, const float* __restrict__ Drow,
                    float* __restrict__ attnOut, unsigned short* __restrict__ Obuf) {
  __shared__ __align__(16) unsigned short tile[8][16 * 32];
  int wslot = threadIdx.x >> 5;
  int wid   = (blockIdx.x << 3) + wslot;                // 0..4095
  int lane  = threadIdx.x & 31;
  int m     = lane & 15;
  int half  = lane >> 4;
  int bh    = wid >> 8;
  int q0    = (wid & 255) << 4;
  int b     = bh >> 3;
  int h     = bh & 7;

  const unsigned short* qp    = Qh + (((size_t)bh * SS + q0 + m) << 6);
  const unsigned short* kbase = Kh + ((size_t)bh * SS << 6);
  const unsigned short* vbase = Vt + (((size_t)bh) << 6) * SS;
  unsigned short* lp = &tile[wslot][0];

  Frag qa0, qa1;
  qa0.q[0] = *(const uint4*)(qp + half * 16);
  qa0.q[1] = *(const uint4*)(qp + half * 16 + 8);
  qa1.q[0] = *(const uint4*)(qp + 32 + half * 16);
  qa1.q[1] = *(const uint4*)(qp + 32 + half * 16 + 8);

  float mr[8], idn[8];
#pragma unroll
  for (int r = 0; r < 8; ++r) {
    int row = q0 + r + (half << 3);
    mr[r]  = Mrow[(size_t)bh * SS + row];
    idn[r] = Drow[(size_t)bh * SS + row];
  }

  v8f o[4] = {{}, {}, {}, {}};
  float* ap_out = attnOut + ((size_t)bh * SS + q0) * SS;

  for (int kb = 0; kb < SS; kb += 32) {
    const unsigned short* k0 = kbase + ((size_t)(kb + m) << 6);
    const unsigned short* k1 = kbase + ((size_t)(kb + 16 + m) << 6);
    __builtin_prefetch(kbase + ((size_t)((kb + 32) & (SS - 1)) << 6), 0, 3);
    Frag b00, b01, b10, b11;
    b00.q[0] = *(const uint4*)(k0 + half * 16);
    b00.q[1] = *(const uint4*)(k0 + half * 16 + 8);
    b01.q[0] = *(const uint4*)(k0 + 32 + half * 16);
    b01.q[1] = *(const uint4*)(k0 + 32 + half * 16 + 8);
    b10.q[0] = *(const uint4*)(k1 + half * 16);
    b10.q[1] = *(const uint4*)(k1 + half * 16 + 8);
    b11.q[0] = *(const uint4*)(k1 + 32 + half * 16);
    b11.q[1] = *(const uint4*)(k1 + 32 + half * 16 + 8);

    v8f c0 = {}; c0 = wmma_bf16(qa0, b00, c0); c0 = wmma_bf16(qa1, b01, c0);
    v8f c1 = {}; c1 = wmma_bf16(qa0, b10, c1); c1 = wmma_bf16(qa1, b11, c1);

#pragma unroll
    for (int r = 0; r < 8; ++r) {
      int row = r + (half << 3);
      float p0 = __expf(c0[r] * 0.125f - mr[r]) * idn[r];
      float p1 = __expf(c1[r] * 0.125f - mr[r]) * idn[r];
      ap_out[(size_t)row * SS + kb + m]      = p0;      // attn output (f32)
      ap_out[(size_t)row * SS + kb + 16 + m] = p1;
      lp[row * 32 + m]      = f2bf(p0);                 // stash for transpose
      lp[row * 32 + 16 + m] = f2bf(p1);
    }

    asm volatile("s_wait_dscnt 0" ::: "memory");        // per-wave LDS RAW fence

    Frag pa;                                            // attn tile as A-frag
    pa.q[0] = *(const uint4*)(lp + m * 32 + half * 16);
    pa.q[1] = *(const uint4*)(lp + m * 32 + half * 16 + 8);

#pragma unroll
    for (int t = 0; t < 4; ++t) {                       // V^T B-frags, dh = t*16+m
      const unsigned short* vp = vbase + (size_t)(t * 16 + m) * SS + kb + half * 16;
      Frag vb;
      vb.q[0] = *(const uint4*)vp;
      vb.q[1] = *(const uint4*)(vp + 8);
      o[t] = wmma_bf16(pa, vb, o[t]);
    }
  }

#pragma unroll
  for (int t = 0; t < 4; ++t)
#pragma unroll
    for (int r = 0; r < 8; ++r) {
      int row = q0 + r + (half << 3);
      Obuf[((size_t)(b * SS + row) << 9) + h * 64 + t * 16 + m] = f2bf(o[t][r]);
    }
}

// ---------------------------------------------------------------------------
// Kernel 4: output projection  out = O @ Wo^T + bo  (all-bf16 operands, f32 out)
// Each wave: 16 rows x 64 cols, 4 accumulators.
// ---------------------------------------------------------------------------
__global__ __launch_bounds__(256)
void out_proj_kernel(const unsigned short* __restrict__ Obuf,
                     const unsigned short* __restrict__ Wob,
                     const float* __restrict__ bo,
                     float* __restrict__ out) {
  int wid  = (blockIdx.x << 3) + (threadIdx.x >> 5);    // 0..4095
  int lane = threadIdx.x & 31;
  int m    = lane & 15;
  int half = lane >> 4;
  int rowt = wid >> 3;                                  // 0..511
  int cg   = wid & 7;
  int r0   = rowt << 4;

  const unsigned short* ap = Obuf + ((size_t)(r0 + m) << 9);

  v8f acc[4] = {{}, {}, {}, {}};
  for (int kb = 0; kb < DD; kb += 32) {
    Frag a;
    a.q[0] = *(const uint4*)(ap + kb + half * 16);
    a.q[1] = *(const uint4*)(ap + kb + half * 16 + 8);
#pragma unroll
    for (int t = 0; t < 4; ++t) {
      int col = (cg << 6) + (t << 4) + m;
      const unsigned short* wp = Wob + ((size_t)col << 9) + kb + half * 16;
      Frag b;
      b.q[0] = *(const uint4*)wp;
      b.q[1] = *(const uint4*)(wp + 8);
      acc[t] = wmma_bf16(a, b, acc[t]);
    }
  }

#pragma unroll
  for (int t = 0; t < 4; ++t) {
    int col  = (cg << 6) + (t << 4) + m;
    float bv = bo[col];
#pragma unroll
    for (int r = 0; r < 8; ++r)
      out[((size_t)(r0 + r + (half << 3)) << 9) + col] = acc[t][r] + bv;
  }
}

// ---------------------------------------------------------------------------
extern "C" void kernel_launch(void* const* d_in, const int* in_sizes, int n_in,
                              void* d_out, int out_size, void* d_ws, size_t ws_size,
                              hipStream_t stream) {
  const float* q   = (const float*)d_in[0];
  const float* k   = (const float*)d_in[1];
  const float* v   = (const float*)d_in[2];
  const float* Wq  = (const float*)d_in[3];
  const float* bq  = (const float*)d_in[4];
  const float* Wk  = (const float*)d_in[5];
  const float* bk  = (const float*)d_in[6];
  const float* Wv  = (const float*)d_in[7];
  const float* bv  = (const float*)d_in[8];
  const float* Wo  = (const float*)d_in[9];
  const float* bo  = (const float*)d_in[10];
  const float* cst = (const float*)d_in[11];

  char* ws = (char*)d_ws;
  const size_t MB = (size_t)1 << 20;
  unsigned short* Xq  = (unsigned short*)(ws);            //  8 MB  bf16 activations
  unsigned short* Xk  = (unsigned short*)(ws + 8 * MB);
  unsigned short* Xv  = (unsigned short*)(ws + 16 * MB);
  unsigned short* Qh  = (unsigned short*)(ws + 24 * MB);  //  8 MB  head-major
  unsigned short* Kh  = (unsigned short*)(ws + 32 * MB);
  unsigned short* Vt  = (unsigned short*)(ws + 40 * MB);  //  transposed V
  unsigned short* Ob  = (unsigned short*)(ws + 48 * MB);
  unsigned short* Wqb = (unsigned short*)(ws + 56 * MB);  // 512 KB each
  unsigned short* Wkb = (unsigned short*)(ws + 56 * MB + 512 * 1024);
  unsigned short* Wvb = (unsigned short*)(ws + 57 * MB);
  unsigned short* Wob = (unsigned short*)(ws + 57 * MB + 512 * 1024);
  float* Mrow = (float*)(ws + 58 * MB);                   // 256 KB
  float* Drow = (float*)(ws + 58 * MB + 256 * 1024);      // 256 KB

  float* out     = (float*)d_out;
  float* attnOut = out + (size_t)BB * SS * DD;            // tuple order: (out, attn)

  dim3 blk(256);
  const int ACT_BLKS = (BB * SS * DD) / (256 * 8);        // 2048
  const int W_BLKS   = (DD * DD) / (256 * 8);             // 128

  // Stage 0: one-time f32 -> bf16 conversions
  cvt_bf16_kernel<<<ACT_BLKS, blk, 0, stream>>>(q, Xq);
  cvt_bf16_kernel<<<ACT_BLKS, blk, 0, stream>>>(k, Xk);
  cvt_bf16_kernel<<<ACT_BLKS, blk, 0, stream>>>(v, Xv);
  cvt_bf16_kernel<<<W_BLKS, blk, 0, stream>>>(Wq, Wqb);
  cvt_bf16_kernel<<<W_BLKS, blk, 0, stream>>>(Wk, Wkb);
  cvt_bf16_kernel<<<W_BLKS, blk, 0, stream>>>(Wv, Wvb);
  cvt_bf16_kernel<<<W_BLKS, blk, 0, stream>>>(Wo, Wob);

  // Stage 1: QKV projections (WMMA GEMM, pure bf16 operands)
  proj_gemm_kernel<<<512, blk, 0, stream>>>(Xq, Wqb, bq, Qh, 0);
  proj_gemm_kernel<<<512, blk, 0, stream>>>(Xk, Wkb, bk, Kh, 0);
  proj_gemm_kernel<<<512, blk, 0, stream>>>(Xv, Wvb, bv, Vt, 1);

  // Stage 2: softmax stats (online max/sum with leaky denominator)
  attn_stats_kernel<<<512, blk, 0, stream>>>(Qh, Kh, cst, Mrow, Drow);

  // Stage 3: attn write + fused attn @ V
  attn_av_kernel<<<512, blk, 0, stream>>>(Qh, Kh, Vt, Mrow, Drow, attnOut, Ob);

  // Stage 4: output projection
  out_proj_kernel<<<512, blk, 0, stream>>>(Ob, Wob, bo, out);
}
